// NSAMSA_360777253457
// MI455X (gfx1250) — matrix operand order repaired and checked by reading, not compiled, not run on hardware
//
#include <hip/hip_runtime.h>
#include <hip/hip_bf16.h>

// Problem constants (from reference)
#define DIM   256
#define NHEAD 8
#define EDIM  32        // per-head dim
#define MBALL 32        // points per ball
#define NBALL 256
#define NMQ   (NBALL * MBALL)   // 8192 queries
#define TOPK  2

typedef __attribute__((ext_vector_type(16))) _Float16 v16h;
typedef __attribute__((ext_vector_type(8)))  float    v8f;

// ---------------------------------------------------------------------------
// Kernel A: per-ball mean of x -> f16 kmean[n][DIM]. Fully coalesced.
// ---------------------------------------------------------------------------
__global__ __launch_bounds__(256) void kmean_kernel(const float* __restrict__ x,
                                                    _Float16* __restrict__ kmeanH) {
    const int ball = blockIdx.x;     // 0..255
    const int t    = threadIdx.x;    // 0..255 (column)
    const float* base = x + (size_t)ball * MBALL * DIM + t;
    float s = 0.f;
#pragma unroll
    for (int m = 0; m < MBALL; ++m) s += base[m * DIM];
    kmeanH[ball * DIM + t] = (_Float16)(s * (1.0f / (float)MBALL));
}

// ---------------------------------------------------------------------------
// Kernel B: one wave per (head, 16-query tile).
//   logits(16x256) = Q(16x32) x KmeanT(32x256) via v_wmma_f32_16x16x32_f16
//   (K=32 -> one WMMA per 16x16 tile; 16 tiles, unrolled x4 for load/WMMA
//   overlap), then per-query stable top-2 + softmax values over 256 balls,
//   scanned by lane pairs with a shfl_xor(16) merge (branchless v_cndmask).
// A-layout (16-bit A 16x32, ISA 7.12.2): lane L holds row M=L&15;
//   half t -> K = (t<8 ? 0:16) + (L>=16 ? 8:0) + 2*((t>>1)&3) + (t&1)
// B-layout (16-bit B 32x16): lane L holds col N=L&15; half t -> K = (t<8 ? t : t+8) + 8*hi (same as A)
// C-layout (f32 16x16): elem v -> row M = v + 8*hi, col N = L&15
// ---------------------------------------------------------------------------
#define LDS_STRIDE 260   // 260 % 64 == 4 -> conflict-light column scans

__global__ __launch_bounds__(32) void sim_top2_kernel(const float* __restrict__ x,
                                                      const _Float16* __restrict__ kmeanH,
                                                      int* __restrict__ topi,
                                                      float* __restrict__ topv) {
    __shared__ float lds[16 * LDS_STRIDE];

    const int blk  = blockIdx.x;          // 0 .. 8*512-1
    const int h    = blk >> 9;            // head
    const int tile = blk & 511;           // 16-query tile within head
    const int q0   = tile * 16;
    const int lane = threadIdx.x;         // 0..31
    const int hi   = lane >> 4;           // 0 / 1
    const int li   = lane & 15;

    // Build A operand: rows = 16 queries of this tile (f32 -> f16 convert).
    v16h a;
#pragma unroll
    for (int t = 0; t < 16; ++t) {
        const int v = t >> 1, r = t & 1;
        const int K = ((v < 4) ? 0 : 16) + (hi ? 8 : 0) + ((v & 3) << 1) + r;
        a[t] = (_Float16)x[(q0 + li) * DIM + h * EDIM + K];
    }

    const float scale = 0.0625f;          // DIM^-0.5 = 1/16

    // 16 ball-tiles of 16 balls each; unroll x4 so B loads for later tiles
    // issue while earlier WMMAs execute.
#pragma unroll 4
    for (int bt = 0; bt < 16; ++bt) {
        v16h b;
#pragma unroll
        for (int t = 0; t < 16; ++t) {
            const int K = ((t < 8) ? t : t + 8) + (hi ? 8 : 0);
            const int n = bt * 16 + li;   // ball (column)
            b[t] = kmeanH[n * DIM + h * EDIM + K];
        }
        v8f c = {};
        c = __builtin_amdgcn_wmma_f32_16x16x32_f16(false, a, false, b,
                                                   (short)0, c, false, false);
#pragma unroll
        for (int v = 0; v < 8; ++v) {
            const int row = v + (hi ? 8 : 0);
            lds[row * LDS_STRIDE + bt * 16 + li] = c[v] * scale;
        }
    }
    __syncthreads();   // single-wave WG: cheap

    // ---- per-query top-2 + softmax denominator, 2 lanes per query ----
    // Lane (li, hi): query q0+li, scans ball indices [hi*128, hi*128+128).
    const float* row = &lds[li * LDS_STRIDE + hi * 128];
    float m1 = -3.402823466e38f, m2 = -3.402823466e38f;
    int   i1 = 0, i2 = 0;
#pragma unroll 4
    for (int n = 0; n < 128; ++n) {
        const float l  = row[n];
        const int  idx = hi * 128 + n;
        const bool g1 = l > m1;
        const bool g2 = l > m2;
        m2 = g1 ? m1 : (g2 ? l   : m2);
        i2 = g1 ? i1 : (g2 ? idx : i2);
        m1 = g1 ? l   : m1;
        i1 = g1 ? idx : i1;
    }
    // Merge the two half-scans (lower-index-first on ties == stable top_k).
    const float om1 = __shfl_xor(m1, 16, 32);
    const int   oi1 = __shfl_xor(i1, 16, 32);
    const float om2 = __shfl_xor(m2, 16, 32);
    const int   oi2 = __shfl_xor(i2, 16, 32);
    const bool selfBest = (m1 > om1) || (m1 == om1 && i1 < oi1);
    const float M1 = selfBest ? m1 : om1;
    const int   I1 = selfBest ? i1 : oi1;
    const float ca = selfBest ? m2 : om2;   // runner-up of winning side
    const int   cia = selfBest ? i2 : oi2;
    const float cb = selfBest ? om1 : m1;   // best of losing side
    const int   cib = selfBest ? oi1 : i1;
    const bool aBest = (ca > cb) || (ca == cb && cia < cib);
    const float M2 = aBest ? ca : cb;
    const int   I2 = aBest ? cia : cib;

    // Softmax denominator, split the same way then combined.
    float z = 0.f;
#pragma unroll 4
    for (int n = 0; n < 128; ++n) z += __expf(row[n] - M1);
    z += __shfl_xor(z, 16, 32);

    if (lane < 16) {
        const int q = q0 + lane;
        const int base = (h * NMQ + q) * TOPK;
        topi[base]     = I1;
        topi[base + 1] = I2;
        topv[base]     = 1.0f / z;              // exp(M1-M1)/z
        topv[base + 1] = __expf(M2 - M1) / z;
    }
}

// ---------------------------------------------------------------------------
// Kernel C: one wave per (head, query); lane owns one head-dim.
// Online softmax over the 64 gathered keys; v == k so the value row is the
// key row already held in the lane register. Coalesced 128B row loads.
// ---------------------------------------------------------------------------
__global__ __launch_bounds__(256) void attn_kernel(const float* __restrict__ x,
                                                   const int* __restrict__ topi,
                                                   const float* __restrict__ topv,
                                                   float* __restrict__ out) {
    const int gwave = (int)((blockIdx.x * blockDim.x + threadIdx.x) >> 5);
    const int lane  = threadIdx.x & 31;
    const int h     = gwave >> 13;        // NMQ = 8192 queries per head
    const int q     = gwave & (NMQ - 1);

    const int base = (h * NMQ + q) * TOPK;
    int   balls[TOPK];
    bool  msk[TOPK];
#pragma unroll
    for (int j = 0; j < TOPK; ++j) {
        balls[j] = topi[base + j];
        msk[j]   = topv[base + j] > 1e-10f;   // wave-uniform
    }
    // Warm the selected balls' first lines (global_prefetch_b8).
#pragma unroll
    for (int j = 0; j < TOPK; ++j)
        __builtin_prefetch(&x[(size_t)(balls[j] * MBALL) * DIM + h * EDIM + lane], 0, 0);

    const float qd = x[q * DIM + h * EDIM + lane] * 0.0625f;

    float runMax = -3.402823466e38f, runSum = 0.f, acc = 0.f;
#pragma unroll
    for (int j = 0; j < TOPK; ++j) {
        if (!msk[j]) continue;                // uniform branch
        const int r0 = balls[j] * MBALL;
#pragma unroll 4
        for (int m = 0; m < MBALL; ++m) {
            const float kd = x[(r0 + m) * DIM + h * EDIM + lane];
            float p = qd * kd;
#pragma unroll
            for (int o = 16; o > 0; o >>= 1) p += __shfl_xor(p, o, 32);
            const float nm = fmaxf(runMax, p);
            const float f  = __expf(runMax - nm);
            const float w  = __expf(p - nm);
            runSum = runSum * f + w;
            acc    = acc * f + w * kd;        // v == k
            runMax = nm;
        }
    }
    out[q * DIM + h * EDIM + lane] = acc / runSum;
}

// ---------------------------------------------------------------------------
// Host-side launcher
// ---------------------------------------------------------------------------
extern "C" void kernel_launch(void* const* d_in, const int* in_sizes, int n_in,
                              void* d_out, int out_size, void* d_ws, size_t ws_size,
                              hipStream_t stream) {
    const float* x = (const float*)d_in[0];
    // d_in[1] (pos) is dead code in the reference.
    float* out = (float*)d_out;

    // Workspace carve-up (~1.13 MB total)
    char* ws = (char*)d_ws;
    _Float16* kmeanH = (_Float16*)ws;                                   // 128 KB
    int*      topi   = (int*)(ws + 131072);                             // 512 KB
    float*    topv   = (float*)(ws + 131072 + 524288);                  // 512 KB

    // 1) per-ball, per-column mean (f16)
    kmean_kernel<<<NBALL, 256, 0, stream>>>(x, kmeanH);

    // 2) WMMA logits + top-2 + softmax values: 8 heads * 512 tiles, 1 wave each
    sim_top2_kernel<<<NHEAD * (NMQ / 16), 32, 0, stream>>>(x, kmeanH, topi, topv);

    // 3) gathered attention: 8*8192 waves, 8 waves per 256-thread block
    attn_kernel<<<(NHEAD * NMQ) / 8, 256, 0, stream>>>(x, topi, topv, out);
}
